// TemporalPyGGraphLayer_16054587752809
// MI455X (gfx1250) — compile-verified, matrix operand order checked by reading, not run on hardware
//
#include <hip/hip_runtime.h>
#include <hip/hip_bf16.h>

// ---------------------------------------------------------------------------
// GAT layer for MI455X (gfx1250, wave32, WMMA).
// Phases:
//   0) memset: out=0, amax=0(encoded -huge), denom=0
//   1) WMMA bf16 GEMM: h[N,128] = x[N,128] @ W[128,128]   (v_wmma_f32_16x16x32_bf16)
//      - W pre-swizzled in LDS into per-lane fragment order (ds_load_b128 path)
//      - A fragments built from 2 contiguous 16B global loads per k-run
//   2) logits: a_src[N,4], a_dst[N,4]
//   3) edge pass A: segment max via atomicMax on order-preserving uint encoding
//   4) edge pass B: e = exp(lrelu - max); atomicAdd denom + weighted scatter of h[src]
//   5) node pass: out /= (denom + 1e-16), + bias
// ---------------------------------------------------------------------------

typedef __attribute__((ext_vector_type(16))) __bf16 bf16x16;
typedef __attribute__((ext_vector_type(8)))  float  f32x8;
typedef __attribute__((ext_vector_type(4)))  float  f32x4;

#define DD 128        // feature dim
#define HH 4          // heads
#define OO 32         // per-head dim
#define TT 4096       // nodes per graph (reference constant)
#define NEG_SLOPE 0.2f

// -------------------- order-preserving float <-> uint encoding -------------
__device__ __forceinline__ unsigned enc_f32(float f) {
  unsigned u = __float_as_uint(f);
  return (u & 0x80000000u) ? ~u : (u | 0x80000000u);
}
__device__ __forceinline__ float dec_f32(unsigned u) {
  unsigned b = (u & 0x80000000u) ? (u & 0x7FFFFFFFu) : ~u;
  return __uint_as_float(b);
}

// -------------------- edge decode (tiled batch + self loops) ----------------
__device__ __forceinline__ void edge_nodes(long long e, const int* __restrict__ ei,
                                           int E, long long BE, int& s, int& d) {
  if (e < BE) {
    long long r = e % E;
    int off = (int)(e / E) * TT;
    s = ei[r] + off;
    d = ei[E + r] + off;
  } else {
    s = d = (int)(e - BE);
  }
}

// ============================================================================
// 1) WMMA GEMM: h = x @ W  (bf16 inputs, f32 accumulate)
//    block = 256 threads = 8 waves; wave => 16 rows x 128 cols; block => 128 rows
//    LDS holds W pre-swizzled to WMMA B-fragment order:
//      Wswz[kt4(4)][t(8)][lane(32)] = 16 contiguous bf16 (32B) per lane
// ============================================================================
__global__ __launch_bounds__(256) void gat_gemm_wmma(
    const float* __restrict__ x, const float* __restrict__ W,
    float* __restrict__ h) {
  __shared__ bf16x16 Wswz[4 * 8 * 32];   // 32 KB of the 320 KB WGP LDS

  const int tid = threadIdx.x;

  // ---- stage W pre-swizzled into fragment order (once per block) ----
  {
    __bf16* wbase = (__bf16*)Wswz;
    for (int slot = tid; slot < 1024; slot += 256) {
      const int kt4 = slot >> 8;          // k tile (0..3)
      const int t   = (slot >> 5) & 7;    // column tile (0..7)
      const int ln  = slot & 31;          // consuming lane
      const int col = t * 16 + (ln & 15);
      const int kb  = kt4 * 32 + ((ln >> 4) << 3);
      __bf16* dst = wbase + (size_t)slot * 16;
#pragma unroll
      for (int i = 0; i < 8; ++i) {
        const int k = kb + 2 * (i & 3) + ((i >> 2) << 4);
        dst[2 * i]     = (__bf16)W[k * DD + col];
        dst[2 * i + 1] = (__bf16)W[(k + 1) * DD + col];
      }
    }
  }
  __syncthreads();

  const int lane = tid & 31;
  const int wave = tid >> 5;
  const int row0 = blockIdx.x * 128 + wave * 16;
  const int m    = lane & 15;
  const int half = lane >> 4;

  f32x8 acc[8] = {};   // 8 column tiles of 16, 64 VGPRs of accumulators

  const float* __restrict__ xrow = x + (size_t)(row0 + m) * DD;

  for (int kt = 0; kt < DD; kt += 32) {
    // A fragment: 16x32 bf16. For this lane it is two contiguous 8-float
    // runs of the row: [kt+8*half .. +7] and [kt+16+8*half .. +7].
    const int base = kt + (half << 3);
    const f32x4 r0 = *(const f32x4*)(xrow + base);
    const f32x4 r1 = *(const f32x4*)(xrow + base + 4);
    const f32x4 r2 = *(const f32x4*)(xrow + base + 16);
    const f32x4 r3 = *(const f32x4*)(xrow + base + 20);
    bf16x16 a;
#pragma unroll
    for (int j = 0; j < 4; ++j) {
      a[j]      = (__bf16)r0[j];
      a[4 + j]  = (__bf16)r1[j];
      a[8 + j]  = (__bf16)r2[j];
      a[12 + j] = (__bf16)r3[j];
    }

    const int ktbase = (kt >> 5) * 256 + lane;   // Wswz[kt4][t][lane]
#pragma unroll
    for (int t = 0; t < 8; ++t) {
      const bf16x16 b = Wswz[ktbase + t * 32];   // 32B contiguous per lane
      acc[t] = __builtin_amdgcn_wmma_f32_16x16x32_bf16(
          /*neg_a=*/false, a, /*neg_b=*/false, b,
          /*c_mod=*/(short)0, acc[t], /*reuse_a=*/false, /*reuse_b=*/false);
    }
  }

  // D tile layout: VGPR r -> row (r + 8*half), lanes 0..15 -> col
#pragma unroll
  for (int t = 0; t < 8; ++t) {
#pragma unroll
    for (int r = 0; r < 8; ++r) {
      const int row = row0 + r + (half << 3);
      const int col = t * 16 + m;
      h[(size_t)row * DD + col] = acc[t][r];
    }
  }
}

// ============================================================================
// 2) per-node attention logits
// ============================================================================
__global__ __launch_bounds__(256) void gat_logits(
    const float* __restrict__ h, const float* __restrict__ att_src,
    const float* __restrict__ att_dst, float* __restrict__ a_src,
    float* __restrict__ a_dst, int N) {
  const int i = blockIdx.x * 256 + threadIdx.x;   // over N*HH
  if (i >= N * HH) return;
  const int n = i >> 2, hd = i & 3;
  const float* __restrict__ hp = h + (size_t)n * DD + hd * OO;
  const float* __restrict__ as = att_src + hd * OO;
  const float* __restrict__ ad = att_dst + hd * OO;
  float s = 0.f, d = 0.f;
#pragma unroll 8
  for (int o = 0; o < OO; ++o) {
    const float v = hp[o];
    s += v * as[o];
    d += v * ad[o];
  }
  a_src[i] = s;
  a_dst[i] = d;
}

// ============================================================================
// 3) edge pass A: segment max (thread per edge*head)
// ============================================================================
__global__ __launch_bounds__(256) void gat_edge_max(
    const int* __restrict__ ei, const float* __restrict__ a_src,
    const float* __restrict__ a_dst, unsigned* __restrict__ amax,
    long long Et, long long BE, int E) {
  const long long i = (long long)blockIdx.x * 256 + threadIdx.x;
  if (i >= Et * HH) return;
  const long long e = i >> 2;
  const int hd = (int)(i & 3);
  int s, d;
  edge_nodes(e, ei, E, BE, s, d);
  float lg = a_src[(size_t)s * HH + hd] + a_dst[(size_t)d * HH + hd];
  lg = (lg > 0.f) ? lg : NEG_SLOPE * lg;
  atomicMax(&amax[(size_t)d * HH + hd], enc_f32(lg));
}

// ============================================================================
// 4) edge pass B: exp, denom accumulate, weighted scatter (wave per edge)
// ============================================================================
__global__ __launch_bounds__(256) void gat_edge_agg(
    const int* __restrict__ ei, const float* __restrict__ a_src,
    const float* __restrict__ a_dst, const unsigned* __restrict__ amax,
    const float* __restrict__ h, float* __restrict__ denom,
    float* __restrict__ out, long long Et, long long BE, int E) {
  const long long wid = (long long)blockIdx.x * 8 + (threadIdx.x >> 5);
  if (wid >= Et) return;
  const int lane = threadIdx.x & 31;

  int s, d;
  edge_nodes(wid, ei, E, BE, s, d);

  float w[HH];
#pragma unroll
  for (int hd = 0; hd < HH; ++hd) {
    float lg = a_src[(size_t)s * HH + hd] + a_dst[(size_t)d * HH + hd];
    lg = (lg > 0.f) ? lg : NEG_SLOPE * lg;
    w[hd] = __expf(lg - dec_f32(amax[(size_t)d * HH + hd]));
  }
  if (lane < HH) atomicAdd(&denom[(size_t)d * HH + lane], w[lane]);

  const float* __restrict__ hs = h + (size_t)s * DD;
  float* __restrict__ od = out + (size_t)d * DD;
#pragma unroll
  for (int hd = 0; hd < HH; ++hd) {
    const int col = hd * OO + lane;
    atomicAdd(&od[col], hs[col] * w[hd]);
  }
}

// ============================================================================
// 5) normalize + bias (in place on out)
// ============================================================================
__global__ __launch_bounds__(256) void gat_finalize(
    float* __restrict__ out, const float* __restrict__ denom,
    const float* __restrict__ bias, int N) {
  const long long i = (long long)blockIdx.x * 256 + threadIdx.x;
  if (i >= (long long)N * DD) return;
  const int n = (int)(i >> 7);
  const int col = (int)(i & 127);
  const int hd = col >> 5;
  out[i] = out[i] / (denom[(size_t)n * HH + hd] + 1e-16f) + bias[col];
}

// ============================================================================
extern "C" void kernel_launch(void* const* d_in, const int* in_sizes, int n_in,
                              void* d_out, int out_size, void* d_ws, size_t ws_size,
                              hipStream_t stream) {
  const float* x       = (const float*)d_in[0];
  const int*   ei      = (const int*)d_in[1];    // [2, E], values in [0, T)
  const float* W       = (const float*)d_in[2];
  const float* att_src = (const float*)d_in[3];
  const float* att_dst = (const float*)d_in[4];
  const float* bias    = (const float*)d_in[5];

  const int N  = in_sizes[0] / DD;        // 32768
  const int E  = in_sizes[1] / 2;         // 65536
  const int B  = N / TT;                  // 8
  const long long BE = (long long)B * E;  // tiled edges
  const long long Et = BE + N;            // + self loops

  // workspace layout (floats)
  float*    h      = (float*)d_ws;                   // N*128
  float*    a_src  = h + (size_t)N * DD;             // N*4
  float*    a_dst  = a_src + (size_t)N * HH;         // N*4
  unsigned* amax   = (unsigned*)(a_dst + (size_t)N * HH); // N*4
  float*    denom  = (float*)(amax + (size_t)N * HH);     // N*4
  float*    out    = (float*)d_out;                  // N*128 accumulate in place

  // phase 0: zero accumulators (graph-capture-legal async memsets)
  hipMemsetAsync(out, 0, (size_t)N * DD * sizeof(float), stream);
  hipMemsetAsync(amax, 0, (size_t)N * HH * 2 * sizeof(unsigned), stream); // amax+denom

  // phase 1: WMMA GEMM (256 blocks x 256 threads; wave = 16x128 strip)
  gat_gemm_wmma<<<N / 128, 256, 0, stream>>>(x, W, h);

  // phase 2: logits
  gat_logits<<<(N * HH + 255) / 256, 256, 0, stream>>>(h, att_src, att_dst,
                                                       a_src, a_dst, N);

  // phase 3: segment max
  {
    const long long tot = Et * HH;
    gat_edge_max<<<(unsigned)((tot + 255) / 256), 256, 0, stream>>>(
        ei, a_src, a_dst, amax, Et, BE, E);
  }

  // phase 4: exp + denom + scatter (wave per edge, 8 waves per block)
  gat_edge_agg<<<(unsigned)((Et + 7) / 8), 256, 0, stream>>>(
      ei, a_src, a_dst, amax, h, denom, out, Et, BE, E);

  // phase 5: normalize + bias
  gat_finalize<<<(unsigned)(((long long)N * DD + 255) / 256), 256, 0, stream>>>(
      out, denom, bias, N);
}
